// MoELayer_79517024518945
// MI455X (gfx1250) — compile-verified, matrix operand order checked by reading, not run on hardware
//
#include <hip/hip_runtime.h>
#include <hip/hip_bf16.h>

// Problem constants (reference: B=4, S=8192, D=512, E=8, K=2)
#define NTOK   (4 * 8192)   // 32768 tokens
#define DDIM   512
#define NEXP   8

// Tiling for the expert GEMM
#define BM 128
#define BN 128
#define KT 64               // K-chunk staged in LDS (= two WMMA k-steps)
#define LDW 72              // LDS row stride in shorts (64 + 8 pad, 16B aligned)

typedef __attribute__((ext_vector_type(16))) __bf16 v16bf;
typedef __attribute__((ext_vector_type(8)))  float  v8f;

union FragAB { v16bf v; uint4 q[2]; };
union Acc8   { v8f v; float f[8]; };

// Packed fp32x2 -> bf16x2 (lowers to v_cvt_pk_bf16_f32 where available)
__device__ __forceinline__ unsigned int pack_bf16x2(float a, float b) {
    union { __hip_bfloat162 h; unsigned int u; } c;
    c.h = __float22bfloat162_rn(make_float2(a, b));
    return c.u;
}

// ---------------- Stage 1: gate -> per-token top-2 softmax weight ----------------
__global__ __launch_bounds__(256)
void MoE_gate_kernel(const float* __restrict__ x,
                     const float* __restrict__ Wg,
                     const float* __restrict__ bg,
                     float* __restrict__ wout) {
    const int wave = threadIdx.x >> 5;
    const int lane = threadIdx.x & 31;
    const int tok  = blockIdx.x * 8 + wave;
    const float* xr = x + (size_t)tok * DDIM;

    float acc[NEXP];
#pragma unroll
    for (int e = 0; e < NEXP; ++e) acc[e] = 0.f;

    for (int d = lane; d < DDIM; d += 32) {
        float xv = xr[d];
        const float4* wg4 = (const float4*)(Wg + (size_t)d * NEXP);
        float4 a = wg4[0], b = wg4[1];
        acc[0] += xv * a.x; acc[1] += xv * a.y; acc[2] += xv * a.z; acc[3] += xv * a.w;
        acc[4] += xv * b.x; acc[5] += xv * b.y; acc[6] += xv * b.z; acc[7] += xv * b.w;
    }
#pragma unroll
    for (int e = 0; e < NEXP; ++e) {
#pragma unroll
        for (int off = 16; off > 0; off >>= 1)
            acc[e] += __shfl_xor(acc[e], off, 32);
    }
    if (lane == 0) {
        float lg[NEXP];
#pragma unroll
        for (int e = 0; e < NEXP; ++e) lg[e] = acc[e] + bg[e];
        float m1 = -1e30f, m2 = -1e30f;
#pragma unroll
        for (int e = 0; e < NEXP; ++e) {
            float v = lg[e];
            if (v > m1) { m2 = m1; m1 = v; }
            else if (v > m2) { m2 = v; }
        }
        float s = 0.f;
#pragma unroll
        for (int e = 0; e < NEXP; ++e) s += __expf(lg[e] - m1);
        wout[tok] = (1.f + __expf(m2 - m1)) / s;
    }
}

// ---------------- Stage 2: out = w * sum_e relu(X @ We[e] + be[e]) ----------------
// Block: 256 threads = 8 waves, tile BM x BN = 128 x 128.
// Wave (wm,wn) in 4x2 grid handles 32 rows x 64 cols = 2x4 WMMA 16x16 frags.
__global__ __launch_bounds__(256)
void MoE_expert_kernel(const float* __restrict__ x,
                       const float* __restrict__ We,
                       const float* __restrict__ be,
                       const float* __restrict__ wrow,
                       float* __restrict__ out) {
    __shared__ __align__(16) unsigned short Xs[BM * LDW];  // [row][k]  bf16 bits
    __shared__ __align__(16) unsigned short Ws[BN * LDW];  // [n][k]    bf16 bits

    const int m0   = blockIdx.y * BM;
    const int n0   = blockIdx.x * BN;
    const int wave = threadIdx.x >> 5;
    const int lane = threadIdx.x & 31;
    const int wm   = wave >> 1;          // 0..3 -> row offset wm*32
    const int wn   = wave & 1;           // 0..1 -> col offset wn*64
    const int l15  = lane & 15;
    const int khalf = (lane >> 4) * 8;   // fragment K base per ISA 16-bit layout

    Acc8 Ssum[2][4];
#pragma unroll
    for (int mf = 0; mf < 2; ++mf)
#pragma unroll
        for (int nf = 0; nf < 4; ++nf)
#pragma unroll
            for (int i = 0; i < 8; ++i) Ssum[mf][nf].f[i] = 0.f;

    for (int e = 0; e < NEXP; ++e) {
        const float* Wexp = We + (size_t)e * DDIM * DDIM;

        Acc8 Cacc[2][4];
#pragma unroll
        for (int mf = 0; mf < 2; ++mf)
#pragma unroll
            for (int nf = 0; nf < 4; ++nf)
#pragma unroll
                for (int i = 0; i < 8; ++i) Cacc[mf][nf].f[i] = 0.f;

        for (int k0 = 0; k0 < DDIM; k0 += KT) {
            // ---- stage X chunk [BM x KT] fp32 -> bf16 LDS (row-major [row][k]) ----
            {
                const int row  = threadIdx.x >> 1;       // 0..127
                const int half = threadIdx.x & 1;        // 32 cols each
                const float4* src =
                    (const float4*)(x + (size_t)(m0 + row) * DDIM + k0 + half * 32);
                uint4* dst = (uint4*)&Xs[row * LDW + half * 32];
#pragma unroll
                for (int v = 0; v < 4; ++v) {
                    float4 f0 = src[2 * v];
                    float4 f1 = src[2 * v + 1];
                    uint4 pk;
                    pk.x = pack_bf16x2(f0.x, f0.y);
                    pk.y = pack_bf16x2(f0.z, f0.w);
                    pk.z = pack_bf16x2(f1.x, f1.y);
                    pk.w = pack_bf16x2(f1.z, f1.w);
                    dst[v] = pk;
                }
            }
            // ---- stage W chunk [KT x BN] fp32 -> transposed bf16 LDS [n][k] ----
            // Pack (k, k+1) pairs so each LDS store is a 32-bit ds_store_b32.
#pragma unroll
            for (int i = 0; i < 16; ++i) {
                const int idx = threadIdx.x + i * 256;   // 0..4095
                const int kp  = idx >> 7;                // 0..31 (k pair)
                const int n   = idx & 127;               // coalesced global reads
                const float w0 = Wexp[(size_t)(k0 + 2 * kp)     * DDIM + n0 + n];
                const float w1 = Wexp[(size_t)(k0 + 2 * kp + 1) * DDIM + n0 + n];
                *(unsigned int*)&Ws[n * LDW + 2 * kp] = pack_bf16x2(w0, w1);
            }
            // Prefetch next k-chunk of this expert's weights into cache.
            if (k0 + KT < DDIM) {
                const float* pf = Wexp + (size_t)(k0 + KT + (threadIdx.x >> 2)) * DDIM
                                  + n0 + (threadIdx.x & 3) * 32;
                __builtin_prefetch(pf, 0, 3);
            }
            __syncthreads();

            // ---- two WMMA k-steps per staged chunk ----
#pragma unroll
            for (int kk = 0; kk < KT; kk += 32) {
                FragAB a[2], b[4];
#pragma unroll
                for (int mf = 0; mf < 2; ++mf) {
                    const int row = wm * 32 + mf * 16 + l15;
                    a[mf].q[0] = *(const uint4*)&Xs[row * LDW + kk + khalf];
                    a[mf].q[1] = *(const uint4*)&Xs[row * LDW + kk + khalf + 16];
                }
#pragma unroll
                for (int nf = 0; nf < 4; ++nf) {
                    const int n = wn * 64 + nf * 16 + l15;
                    b[nf].q[0] = *(const uint4*)&Ws[n * LDW + kk + khalf];
                    b[nf].q[1] = *(const uint4*)&Ws[n * LDW + kk + khalf + 16];
                }
#pragma unroll
                for (int mf = 0; mf < 2; ++mf)
#pragma unroll
                    for (int nf = 0; nf < 4; ++nf)
                        Cacc[mf][nf].v = __builtin_amdgcn_wmma_f32_16x16x32_bf16(
                            false, a[mf].v, false, b[nf].v,
                            (short)0, Cacc[mf][nf].v, false, false);
            }
            __syncthreads();
        }

        // ---- bias + relu, accumulate into running sum ----
#pragma unroll
        for (int nf = 0; nf < 4; ++nf) {
            const int n = n0 + wn * 64 + nf * 16 + l15;
            const float bias = be[e * DDIM + n];
#pragma unroll
            for (int mf = 0; mf < 2; ++mf)
#pragma unroll
                for (int i = 0; i < 8; ++i) {
                    float v = Cacc[mf][nf].f[i] + bias;
                    Ssum[mf][nf].f[i] += (v > 0.f) ? v : 0.f;
                }
        }
    }

    // ---- epilogue: scale by per-token gate weight, write out ----
#pragma unroll
    for (int mf = 0; mf < 2; ++mf) {
        const int mbase = m0 + wm * 32 + mf * 16 + khalf; // khalf = (lane>>4)*8
#pragma unroll
        for (int nf = 0; nf < 4; ++nf) {
            const int n = n0 + wn * 64 + nf * 16 + l15;
#pragma unroll
            for (int i = 0; i < 8; ++i) {
                const int m = mbase + i;
                out[(size_t)m * DDIM + n] = wrow[m] * Ssum[mf][nf].f[i];
            }
        }
    }
}

extern "C" void kernel_launch(void* const* d_in, const int* in_sizes, int n_in,
                              void* d_out, int out_size, void* d_ws, size_t ws_size,
                              hipStream_t stream) {
    const float* x  = (const float*)d_in[0];
    const float* We = (const float*)d_in[1];
    const float* be = (const float*)d_in[2];
    const float* Wg = (const float*)d_in[3];
    const float* bg = (const float*)d_in[4];
    float* out  = (float*)d_out;
    float* wtok = (float*)d_ws;   // NTOK floats of per-token gate weight

    // Stage 1: gate weights (one wave32 per token)
    MoE_gate_kernel<<<NTOK / 8, 256, 0, stream>>>(x, Wg, bg, wtok);

    // Stage 2: fused expert GEMMs + relu-sum + scaling
    dim3 grid(DDIM / BN, NTOK / BM);   // (4, 256)
    MoE_expert_kernel<<<grid, 256, 0, stream>>>(x, We, be, wtok, out);
}
